// Multi_Head_CrossAttention_61572651155707
// MI455X (gfx1250) — matrix operand order repaired and checked by reading, not compiled, not run on hardware
//
#include <hip/hip_runtime.h>

// ---------------------------------------------------------------------------
// Multi-head cross-attention forward for MI455X (gfx1250), wave32 + WMMA.
// Pipeline: 3x projection GEMM (f32->f16, 128x128 blocks) -> flash attention
// (32 queries/wave, f16 WMMA, f32 accum, online softmax via DPP-max +
// ones-WMMA rowsum, dropout additive mask) -> final fc GEMM (f32 out).
// ---------------------------------------------------------------------------

typedef __attribute__((ext_vector_type(16))) _Float16 v16h;
typedef __attribute__((ext_vector_type(2)))  _Float16 v2h;
typedef __attribute__((ext_vector_type(8)))  float    v8f;

#define DEV __device__ __forceinline__

// ---- WMMA fragment helpers (layouts per CDNA5 ISA 7.12.2, wave32) ---------

// A fragment: 16x32 f16, element [m,k] from row-major src (row stride ld halfs).
DEV v16h load_A_frag(const _Float16* __restrict__ src, int ld) {
  const int lane = threadIdx.x & 31;
  const int row  = lane & 15;
  const int kb   = (lane >> 4) * 8;
  const _Float16* p = src + (size_t)row * ld;
  v16h a;
#pragma unroll
  for (int v = 0; v < 4; ++v) {
    a[2 * v]     = p[kb + 2 * v];
    a[2 * v + 1] = p[kb + 2 * v + 1];
  }
#pragma unroll
  for (int v = 0; v < 4; ++v) {
    a[8 + 2 * v]     = p[16 + kb + 2 * v];
    a[8 + 2 * v + 1] = p[16 + kb + 2 * v + 1];
  }
  return a;
}

// B fragment: 32x16 f16, element [k,n] = src[n*ld + k] (contiguous K-run per col).
DEV v16h load_Bt_frag(const _Float16* __restrict__ src, int ld) {
  const int lane = threadIdx.x & 31;
  const int n    = lane & 15;
  const int kb   = (lane >> 4) * 8;
  const _Float16* p = src + (size_t)n * ld + kb;
  v16h b;
#pragma unroll
  for (int i = 0; i < 16; ++i) b[i] = p[i < 8 ? i : i + 8];
  return b;
}

DEV v8f wmma_f16(v16h a, v16h b, v8f c) {
  return __builtin_amdgcn_wmma_f32_16x16x32_f16(false, a, false, b,
                                                (short)0, c, false, false);
}

// ---- 16-lane row max via DPP butterfly (no LDS traffic) -------------------
template <int CTRL>
DEV float fmax_dpp(float x) {
  const int xi = __float_as_int(x);
  const int r  = __builtin_amdgcn_update_dpp(xi, xi, CTRL, 0xF, 0xF, true);
  return fmaxf(x, __int_as_float(r));
}
DEV float rowmax16(float x) {
  x = fmax_dpp<0xB1>(x);    // quad_perm [1,0,3,2] : xor 1
  x = fmax_dpp<0x4E>(x);    // quad_perm [2,3,0,1] : xor 2
  x = fmax_dpp<0x141>(x);   // row_half_mirror     : xor 7
  x = fmax_dpp<0x140>(x);   // row_mirror          : xor 15
  return x;
}

// intra-32 key permutation: position j holds original key perm(j) = (j&1)*16+(j>>1).
DEV int kperm_pos(int t) { return 2 * (t & 15) + ((t >> 4) & 1); }

// ---------------------------------------------------------------------------
// Tiled GEMM:  C[M,N] = A[M,K] * W[N,K]^T.  256 threads, 128x128 block,
// wave w -> 16-row stripe x 128 cols (8 acc frags). K staged in LDS as f16.
// OUTMODE: 0 = f16 row-major, 1 = f16 transposed [N,M] + intra-32 row perm
//          (for V), 2 = f32 row-major.
// ---------------------------------------------------------------------------
template <typename AT, int OUTMODE>
__global__ __launch_bounds__(256) void gemm_wmma(
    const AT* __restrict__ A, const float* __restrict__ W,
    void* __restrict__ Cout, int M, int N, int K) {
  constexpr int ST = 34;
  __shared__ _Float16 sA[128 * ST];
  __shared__ _Float16 sB[128 * ST];

  const int tid  = threadIdx.x;
  const int wave = tid >> 5;
  const int lane = tid & 31;
  const int m0 = blockIdx.x * 128;
  const int n0 = blockIdx.y * 128;

  const v8f vz = {0.f, 0.f, 0.f, 0.f, 0.f, 0.f, 0.f, 0.f};
  v8f acc[8];
#pragma unroll
  for (int t = 0; t < 8; ++t) acc[t] = vz;

  for (int kk = 0; kk < K; kk += 32) {
    {
      const int k = tid & 31;
#pragma unroll
      for (int r = tid >> 5; r < 128; r += 8) {
        sA[r * ST + k] = (_Float16)A[(size_t)(m0 + r) * K + kk + k];
        sB[r * ST + k] = (_Float16)W[(size_t)(n0 + r) * K + kk + k];
      }
    }
    __syncthreads();

    v16h a = load_A_frag(&sA[wave * 16 * ST], ST);
#pragma unroll
    for (int t = 0; t < 8; ++t) {
      v16h b = load_Bt_frag(&sB[t * 16 * ST], ST);
      acc[t] = wmma_f16(a, b, acc[t]);
    }
    __syncthreads();
  }

  const int col = lane & 15;
  const int rb  = (lane >> 4) * 8;
#pragma unroll
  for (int t = 0; t < 8; ++t) {
#pragma unroll
    for (int r = 0; r < 8; ++r) {
      const int row = m0 + wave * 16 + rb + r;
      const int cg  = n0 + t * 16 + col;
      const float v = acc[t][r];
      if (OUTMODE == 0) {
        ((_Float16*)Cout)[(size_t)row * N + cg] = (_Float16)v;
      } else if (OUTMODE == 1) {
        const int rp = (row & ~31) + kperm_pos(row & 31);
        ((_Float16*)Cout)[(size_t)cg * M + rp] = (_Float16)v;
      } else {
        ((float*)Cout)[(size_t)row * N + cg] = v;
      }
    }
  }
}

// ---------------------------------------------------------------------------
// Flash attention: grid (SQ/128, H), 128 threads. Each wave owns TWO 16-row
// query tiles (32 queries) so K fragments are shared across q-tiles and V
// fragments across both P tiles. Loops over 32-key blocks, online softmax.
// Qf,Kf: [SQ, H*64] f16.  Vt: [H*64, SK] f16 (transposed, intra-32 permuted).
// U: drop_u [H,SQ,SK] f32.
// ---------------------------------------------------------------------------
__global__ __launch_bounds__(128) void attn_wmma(
    const _Float16* __restrict__ Qf, const _Float16* __restrict__ Kf,
    const _Float16* __restrict__ Vt, const float* __restrict__ U,
    _Float16* __restrict__ Aout) {
  constexpr int SQ = 2048, SK = 2048, HD = 1024;
  constexpr int PST = 34;
  __shared__ _Float16 sP[4 * 2 * 16 * PST];   // per-wave, per-q-tile P staging

  const int tid  = threadIdx.x;
  const int wave = tid >> 5;
  const int lane = tid & 31;
  const int col  = lane & 15;
  const int rb   = (lane >> 4) * 8;
  const int h    = blockIdx.y;
  const int q0   = (blockIdx.x * 4 + wave) * 32;
  _Float16* pPt[2] = { &sP[(wave * 2 + 0) * 16 * PST],
                       &sP[(wave * 2 + 1) * 16 * PST] };

  // Q tiles (2 x 16x64) as A-fragments, loaded once
  const _Float16* Qb = Qf + (size_t)q0 * HD + h * 64;
  v16h aq[2][2];
#pragma unroll
  for (int qt = 0; qt < 2; ++qt)
#pragma unroll
    for (int kc = 0; kc < 2; ++kc)
      aq[qt][kc] = load_A_frag(Qb + (size_t)qt * 16 * HD + kc * 32, HD);

  v16h bones;
#pragma unroll
  for (int i = 0; i < 16; ++i) bones[i] = (_Float16)1.0f;

  const v8f vz = {0.f, 0.f, 0.f, 0.f, 0.f, 0.f, 0.f, 0.f};
  v8f acc[2][4];
  float mrow[2][8], lrow[2][8];
#pragma unroll
  for (int qt = 0; qt < 2; ++qt) {
#pragma unroll
    for (int t = 0; t < 4; ++t) acc[qt][t] = vz;
#pragma unroll
    for (int r = 0; r < 8; ++r) { mrow[qt][r] = -INFINITY; lrow[qt][r] = 0.f; }
  }

  for (int k0 = 0; k0 < SK; k0 += 32) {
    // ---- K fragments, shared by both q-tiles -----------------------------
    v16h bk[2][2];
#pragma unroll
    for (int nt = 0; nt < 2; ++nt)
#pragma unroll
      for (int kc = 0; kc < 2; ++kc)
        bk[nt][kc] = load_Bt_frag(
            Kf + (size_t)(k0 + nt * 16) * HD + h * 64 + kc * 32, HD);

    // ---- per q-tile: S, scale+mask, online softmax, stage P --------------
#pragma unroll
    for (int qt = 0; qt < 2; ++qt) {
      float s[2][8];
#pragma unroll
      for (int nt = 0; nt < 2; ++nt) {
        v8f d = vz;
        d = wmma_f16(aq[qt][0], bk[nt][0], d);
        d = wmma_f16(aq[qt][1], bk[nt][1], d);
        const float* Ub = U + ((size_t)h * SQ + q0 + qt * 16 + rb) * SK
                            + k0 + nt * 16 + col;
#pragma unroll
        for (int r = 0; r < 8; ++r) {
          const float u = Ub[(size_t)r * SK];
          s[nt][r] = d[r] * 0.125f + (u < 0.1f ? -1.0e12f : 0.0f);
        }
      }
      _Float16* pP = pPt[qt];
#pragma unroll
      for (int r = 0; r < 8; ++r) {
        const float rm   = rowmax16(fmaxf(s[0][r], s[1][r]));
        const float mn   = fmaxf(mrow[qt][r], rm);
        const float corr = __expf(mrow[qt][r] - mn);
        mrow[qt][r] = mn;
        const _Float16 h0 = (_Float16)__expf(s[0][r] - mn);  // key k0+col
        const _Float16 h1 = (_Float16)__expf(s[1][r] - mn);  // key k0+16+col
        *(v2h*)&pP[(rb + r) * PST + 2 * col] = (v2h){h0, h1};
        lrow[qt][r] *= corr;
        acc[qt][0][r] *= corr; acc[qt][1][r] *= corr;
        acc[qt][2][r] *= corr; acc[qt][3][r] *= corr;
      }
    }

    // ---- P @ V-block: V fragments shared by both P tiles -----------------
    v16h ap0 = load_A_frag(pPt[0], PST);
    v16h ap1 = load_A_frag(pPt[1], PST);
    v8f ds0 = wmma_f16(ap0, bones, vz);
    v8f ds1 = wmma_f16(ap1, bones, vz);
#pragma unroll
    for (int t = 0; t < 4; ++t) {
      v16h bv = load_Bt_frag(Vt + (size_t)(h * 64 + t * 16) * SK + k0, SK);
      acc[0][t] = wmma_f16(ap0, bv, acc[0][t]);
      acc[1][t] = wmma_f16(ap1, bv, acc[1][t]);
    }
#pragma unroll
    for (int r = 0; r < 8; ++r) { lrow[0][r] += ds0[r]; lrow[1][r] += ds1[r]; }
  }

  // ---- normalize and store f16 to [SQ, H*64] -----------------------------
#pragma unroll
  for (int qt = 0; qt < 2; ++qt)
#pragma unroll
    for (int t = 0; t < 4; ++t)
#pragma unroll
      for (int r = 0; r < 8; ++r) {
        const float v = acc[qt][t][r] / lrow[qt][r];
        Aout[(size_t)(q0 + qt * 16 + rb + r) * HD + h * 64 + t * 16 + col] =
            (_Float16)v;
      }
}

// ---------------------------------------------------------------------------
extern "C" void kernel_launch(void* const* d_in, const int* in_sizes, int n_in,
                              void* d_out, int out_size, void* d_ws, size_t ws_size,
                              hipStream_t stream) {
  (void)in_sizes; (void)n_in; (void)out_size; (void)ws_size;
  constexpr int M = 2048, D = 1024, H = 16;

  const float* inQ = (const float*)d_in[0];
  const float* inK = (const float*)d_in[1];
  const float* inV = (const float*)d_in[2];
  const float* WQ  = (const float*)d_in[3];
  const float* WK  = (const float*)d_in[4];
  const float* Wfc = (const float*)d_in[5];
  const float* U   = (const float*)d_in[6];

  // workspace: 4 x (2048*1024) f16 = 16 MB
  _Float16* Qf = (_Float16*)d_ws;
  _Float16* Kf = Qf + (size_t)M * D;
  _Float16* Vt = Kf + (size_t)M * D;
  _Float16* Af = Vt + (size_t)M * D;

  const dim3 gblk(256);
  const dim3 gG(M / 128, D / 128);

  gemm_wmma<float, 0><<<gG, gblk, 0, stream>>>(inQ, WQ, Qf, M, D, D);       // Q
  gemm_wmma<float, 0><<<gG, gblk, 0, stream>>>(inK, WK, Kf, M, D, D);       // K
  gemm_wmma<float, 1><<<gG, gblk, 0, stream>>>(inV, WQ, Vt, M, D, D);       // V (W_Q!)
  attn_wmma<<<dim3(M / 128, H), dim3(128), 0, stream>>>(Qf, Kf, Vt, U, Af); // attention
  gemm_wmma<_Float16, 2><<<gG, gblk, 0, stream>>>(Af, Wfc, d_out, M, D, D); // final fc
}
